// BARNNLSTM_69320772158124
// MI455X (gfx1250) — compile-verified, matrix operand order checked by reading, not run on hardware
//
#include <hip/hip_runtime.h>
#include <cstdint>
#include <cstddef>

// Problem constants (match reference)
#define I_   256
#define H_   512
#define L_   2
#define B_   64
#define T_   2048
#define EH_  128
#define NBLK 16     // persistent grid: 16 WGs x 256 threads = 128 wave32

typedef __bf16 bf16_t;
typedef float    v8f   __attribute__((ext_vector_type(8)));
typedef bf16_t   v16bf __attribute__((ext_vector_type(16)));
typedef unsigned v4u   __attribute__((ext_vector_type(4)));
typedef int      v8i   __attribute__((ext_vector_type(8)));
typedef int      v4i   __attribute__((ext_vector_type(4)));

// ---------------------------------------------------------------- helpers ---
__device__ inline bf16_t f2bf(float f) { return (bf16_t)f; }   // native v_cvt
__device__ inline float  bf2f(bf16_t b) { return (float)b; }

__device__ inline float sigf(float x)  { return 1.f / (1.f + __expf(-x)); }
__device__ inline float leaky(float x) { return x > 0.f ? x : 0.01f * x; }

__device__ inline v8f wmma_bf16(v16bf a, v16bf b, v8f c) {
    // D = A(16x32) * B(32x16) + C, f32 accumulate
    return __builtin_amdgcn_wmma_f32_16x16x32_bf16(
        false, a, false, b, (short)0, c, false, false);
}

// A/B fragment from row-major bf16 [*, ld] (global or LDS).
// Lane l (half h=l>>4): row = row0 + (l&15); elements j0..7 <- K = k0+8h..,
// j8..15 <- K = k0+16+8h.. : two contiguous 16B chunks -> b128 loads.
__device__ inline v16bf load_frag_bf16(const bf16_t* p, int ld, int row0, int k0, int lane) {
    const int r = row0 + (lane & 15);
    const int h = (lane >> 4) & 1;
    const bf16_t* b0 = p + (size_t)r * ld + k0 + 8 * h;
    union { v16bf v; uint4 u[2]; } f;
    f.u[0] = *reinterpret_cast<const uint4*>(b0);
    f.u[1] = *reinterpret_cast<const uint4*>(b0 + 16);
    return f.v;
}

// Tensor Data Mover: DMA a [rows x cols] bf16 tile (row-major, stride=row_stride)
// from global into LDS. One call per wave issues one descriptor; tracked by
// TENSORcnt, in-order completion per wave. Layout per CDNA5 ISA 8.3/8.4.
__device__ inline void tdm_load_tile_bf16(unsigned lds_addr, const bf16_t* gptr,
                                          int cols, int rows, int row_stride) {
    const unsigned long long ga = (unsigned long long)(uintptr_t)gptr;
    v4u g0;
    g0[0] = 1u;                                         // count=1, user descriptor
    g0[1] = lds_addr;                                   // lds_addr [63:32]
    g0[2] = (unsigned)(ga & 0xFFFFFFFFu);               // global_addr [95:64]
    g0[3] = (unsigned)((ga >> 32) & 0x01FFFFFFu)        // global_addr [120:96]
          | (2u << 30);                                 // type=2 ("image")
    v8i g1;
    g1[0] = (int)(1u << 16);                            // data_size=1 (2 bytes)
    g1[1] = (int)(((unsigned)cols & 0xFFFFu) << 16);    // tensor_dim0[15:0] @ bit48
    g1[2] = (int)(((unsigned)rows & 0xFFFFu) << 16);    // tensor_dim1[15:0] @ bit80
    g1[3] = (int)(((unsigned)cols & 0xFFFFu) << 16);    // tile_dim0 @ bit112
    g1[4] = (int)((unsigned)rows & 0xFFFFu);            // tile_dim1 @ bit128
    g1[5] = (int)row_stride;                            // tensor_dim0_stride[31:0] @ bit160
    g1[6] = 0;
    g1[7] = 0;
    v4i gz = {0, 0, 0, 0};
#if __clang_major__ >= 23
    v8i gz8 = {0, 0, 0, 0, 0, 0, 0, 0};
    __builtin_amdgcn_tensor_load_to_lds(g0, g1, gz, gz, gz8, 0);
#else
    __builtin_amdgcn_tensor_load_to_lds(g0, g1, gz, gz, 0);
#endif
}

// ------------------------------------------------------------ init / cvt ----
__global__ __launch_bounds__(256) void k_init(const float* h0, const float* c0,
                                              float* hs, float* cs, unsigned* bar) {
    const int tid = blockIdx.x * blockDim.x + threadIdx.x;
    if (tid == 0) *bar = 0u;
    const int n = L_ * B_ * H_;
    for (int i = tid; i < n; i += gridDim.x * blockDim.x) { hs[i] = h0[i]; cs[i] = c0[i]; }
}

struct CvtArgs {
    const float* src[10];
    bf16_t*      dst[10];
    int          n[10];
};

__global__ __launch_bounds__(256) void k_cvt(CvtArgs a) {
    const int stride = gridDim.x * blockDim.x;
    const int tid    = blockIdx.x * blockDim.x + threadIdx.x;
#pragma unroll
    for (int s = 0; s < 10; ++s) {
        const float* sp = a.src[s];
        bf16_t*      dp = a.dst[s];
        const int    n  = a.n[s];
        for (int i = tid; i < n; i += stride) dp[i] = f2bf(sp[i]);
    }
}

// --------------------------------------------------- encoder_y (parallel) ---
// One 16-row tile of (b*T+t) per workgroup; 4 waves share LDS intermediates.
// chain: x[16,I] -> leaky(W1) [16,EH] -> leaky(W2) [16,EH] -> exp(W3) [16,I]
// then xs[t][b][i] = x[b][t][i] * alpha   (bf16)
__global__ __launch_bounds__(128) void k_ency(const float* __restrict__ x,
                                              const float* eyb1, const float* eyb2, const float* eyb3,
                                              const bf16_t* w1, const bf16_t* w2, const bf16_t* w3,
                                              bf16_t* __restrict__ xs) {
    __shared__ bf16_t xT[16 * I_];     // staged input tile (bf16)
    __shared__ bf16_t e1[16 * EH_];
    __shared__ bf16_t e2[16 * EH_];
    const int wave = threadIdx.x >> 5, lane = threadIdx.x & 31;
    const int row0 = blockIdx.x * 16;           // row in [B*T, I]

    // stage x tile: f32 -> bf16 once (instead of per-fragment conversion)
    for (int i = threadIdx.x; i < 16 * I_; i += 128) {
        const int rr = i >> 8;            // / I_
        const int cc = i & (I_ - 1);
        xT[i] = f2bf(x[(size_t)(row0 + rr) * I_ + cc]);
    }
    __syncthreads();

    // L1: K = I
    for (int nt = wave; nt < EH_ / 16; nt += 4) {
        v8f acc = {};
        for (int ks = 0; ks < I_ / 32; ++ks) {
            v16bf a = load_frag_bf16(xT, I_, 0, ks * 32, lane);
            v16bf b = load_frag_bf16(w1, I_, nt * 16, ks * 32, lane);
            acc = wmma_bf16(a, b, acc);
        }
        const int n = nt * 16 + (lane & 15);
        const float bias = eyb1[n];
        const int rb = 8 * (lane >> 4);
#pragma unroll
        for (int r = 0; r < 8; ++r) e1[(rb + r) * EH_ + n] = f2bf(leaky(acc[r] + bias));
    }
    __syncthreads();

    // L2: K = EH
    for (int nt = wave; nt < EH_ / 16; nt += 4) {
        v8f acc = {};
        for (int ks = 0; ks < EH_ / 32; ++ks) {
            v16bf a = load_frag_bf16(e1, EH_, 0, ks * 32, lane);
            v16bf b = load_frag_bf16(w2, EH_, nt * 16, ks * 32, lane);
            acc = wmma_bf16(a, b, acc);
        }
        const int n = nt * 16 + (lane & 15);
        const float bias = eyb2[n];
        const int rb = 8 * (lane >> 4);
#pragma unroll
        for (int r = 0; r < 8; ++r) e2[(rb + r) * EH_ + n] = f2bf(leaky(acc[r] + bias));
    }
    __syncthreads();

    // L3: alpha = exp(.), scale input, write xs (t-major)
    for (int nt = wave; nt < I_ / 16; nt += 4) {
        v8f acc = {};
        for (int ks = 0; ks < EH_ / 32; ++ks) {
            v16bf a = load_frag_bf16(e2, EH_, 0, ks * 32, lane);
            v16bf b = load_frag_bf16(w3, EH_, nt * 16, ks * 32, lane);
            acc = wmma_bf16(a, b, acc);
        }
        const int n = nt * 16 + (lane & 15);
        const float bias = eyb3[n];
        const int rb = 8 * (lane >> 4);
#pragma unroll
        for (int r = 0; r < 8; ++r) {
            const int gr = row0 + rb + r;        // gr = b*T + t
            const int b_ = gr >> 11;             // T = 2048
            const int t_ = gr & (T_ - 1);
            const float alpha = __expf(acc[r] + bias);
            const float xv = bf2f(xT[(rb + r) * I_ + n]);
            xs[(size_t)t_ * B_ * I_ + (size_t)b_ * I_ + n] = f2bf(xv * alpha);
        }
    }
}

// ------------------------------------------------------- persistent RNN -----
struct RnnArgs {
    unsigned* bar;
    const bf16_t* xs;                     // [T][B][I] bf16
    bf16_t* hsc;                          // [L][B][H] bf16 (scaled h)
    bf16_t* h0n;                          // [B][H] bf16 (layer0 new h)
    const bf16_t* Wih0; const bf16_t* Whh0;
    const bf16_t* Wih1; const bf16_t* Whh1;
    const bf16_t* Ehw1; const bf16_t* Ehw2; const bf16_t* Ehw3;
    const float* bih0; const float* bhh0; const float* bih1; const float* bhh1;
    const float* ehb1; const float* ehb2; const float* ehb3;
    float* out;                           // [B][T][H]
    float* h_state;                       // [L][B][H] (lives in d_out tail)
    float* c_state;                       // [L][B][H] (lives in d_out tail)
};

__device__ inline void gbar(unsigned* cnt, unsigned& epoch) {
    __syncthreads();
    epoch++;
    if (threadIdx.x == 0) {
        __threadfence();
        atomicAdd(cnt, 1u);
        const unsigned target = epoch * NBLK;
        while (__hip_atomic_load(cnt, __ATOMIC_ACQUIRE, __HIP_MEMORY_SCOPE_AGENT) < target) {
            __builtin_amdgcn_s_sleep(2);
        }
        __threadfence();
    }
    __syncthreads();
}

// Accumulate the four gate tiles (i,f,g,o) for one (ht) hidden tile:
// acc[g] += A[16,K] x W[g*H + ht*16 .. +16, K]^T.  A ld == K.
__device__ inline void gemm_quad(v8f acc[4], const bf16_t* Amat, int Arow0, int K,
                                 const bf16_t* W, int ht, int lane) {
    for (int ks = 0; ks < K / 32; ++ks) {
        // prefetch next weight chunk into the cache hierarchy (global_prefetch_b8)
        if (ks + 2 < K / 32)
            __builtin_prefetch(W + (size_t)(ht * 16 + (lane & 15)) * K + (ks + 2) * 32, 0, 0);
        v16bf a = load_frag_bf16(Amat, K, Arow0, ks * 32, lane);
#pragma unroll
        for (int g = 0; g < 4; ++g) {
            v16bf b = load_frag_bf16(W, K, g * H_ + ht * 16, ks * 32, lane);
            acc[g] = wmma_bf16(a, b, acc[g]);
        }
    }
}

// Fused LSTM cell update on the four gate accumulators.
__device__ inline void cell_update(const v8f acc[4], const float bg[4],
                                   int t, int mt, int ht, int lane,
                                   float* hst, float* cst,
                                   bf16_t* hout_bf, float* out_f32) {
    const int hu = ht * 16 + (lane & 15);
    const int rb = 8 * (lane >> 4);
#pragma unroll
    for (int r = 0; r < 8; ++r) {
        const int m = mt * 16 + rb + r;                    // batch index
        const size_t idx = (size_t)m * H_ + hu;
        const float iv = sigf (acc[0][r] + bg[0]);
        const float fv = sigf (acc[1][r] + bg[1]);
        const float gv = tanhf(acc[2][r] + bg[2]);
        const float ov = sigf (acc[3][r] + bg[3]);
        const float cn = fv * cst[idx] + iv * gv;
        const float hn = ov * tanhf(cn);
        cst[idx] = cn;
        hst[idx] = hn;
        if (hout_bf) hout_bf[idx] = f2bf(hn);
        if (out_f32) out_f32[(size_t)m * T_ * H_ + (size_t)t * H_ + hu] = hn;
    }
}

__global__ __launch_bounds__(256) void k_rnn(RnnArgs A) {
    __shared__ bf16_t hT[16 * H_];     // staged h_state tile (bf16)  16 KB
    __shared__ bf16_t e1[16 * EH_];    //                              4 KB
    __shared__ bf16_t e2[16 * EH_];    //                              4 KB
    __shared__ bf16_t xtile[16 * I_];  // TDM: x_scaled[t] tile        8 KB
    __shared__ bf16_t sh0[16 * H_];    // TDM: h_scaled[0] tile       16 KB
    __shared__ bf16_t sh1[16 * H_];    // TDM: h_scaled[1] tile       16 KB
    const int wave  = threadIdx.x >> 5;
    const int lane  = threadIdx.x & 31;
    const int gwave = blockIdx.x * 8 + wave;   // 0..127
    const int mt    = gwave >> 5;              // batch tile (uniform per block)
    const int ht    = gwave & 31;              // hidden tile
    unsigned epoch = 0;
    const unsigned xtile_lds = (unsigned)(uintptr_t)(void*)xtile;
    const unsigned sh0_lds   = (unsigned)(uintptr_t)(void*)sh0;
    const unsigned sh1_lds   = (unsigned)(uintptr_t)(void*)sh1;

    // loop-invariant biases -> registers (off the 2048-step critical path)
    const int hu = ht * 16 + (lane & 15);
    float bgB[4], bgC[4];
#pragma unroll
    for (int g = 0; g < 4; ++g) {
        bgB[g] = A.bih0[g * H_ + hu] + A.bhh0[g * H_ + hu];
        bgC[g] = A.bih1[g * H_ + hu] + A.bhh1[g * H_ + hu];
    }
    const int en = wave * 16 + (lane & 15);    // encoder column for L1/L2
    const float be1 = A.ehb1[en];
    const float be2 = A.ehb2[en];
    float be3[4];
#pragma unroll
    for (int j = 0; j < 4; ++j) be3[j] = A.ehb3[(wave + 8 * j) * 16 + (lane & 15)];

    for (int t = 0; t < T_; ++t) {
        // ---- TDM #1: DMA x_scaled[t] rows [mt*16, mt*16+16) into LDS.
        // Overlaps the phase-A encoder GEMMs; EXEC-ignored, one descriptor
        // per block (wave 0 only), tracked by TENSORcnt (in-order per wave).
        if (wave == 0) {
            tdm_load_tile_bf16(xtile_lds,
                               A.xs + (size_t)t * B_ * I_ + (size_t)mt * 16 * I_,
                               I_, 16, I_);
        }

        // ---- Phase A: alpha_h encoder over h_state [L*B=128, H], 16 rows/WG
        if (blockIdx.x < 8) {
            const int row0 = blockIdx.x * 16;
            // stage h tile f32 -> bf16 once
            for (int i = threadIdx.x; i < 16 * H_; i += 256) {
                const int rr = i >> 9;            // / H_
                const int cc = i & (H_ - 1);
                hT[i] = f2bf(A.h_state[(size_t)(row0 + rr) * H_ + cc]);
            }
            __syncthreads();
            {   // L1: K = H, one n-tile per wave (EH/16 == 8 waves)
                v8f acc = {};
                for (int ks = 0; ks < H_ / 32; ++ks) {
                    v16bf a = load_frag_bf16(hT, H_, 0, ks * 32, lane);
                    v16bf b = load_frag_bf16(A.Ehw1, H_, wave * 16, ks * 32, lane);
                    acc = wmma_bf16(a, b, acc);
                }
                const int rb = 8 * (lane >> 4);
#pragma unroll
                for (int r = 0; r < 8; ++r) e1[(rb + r) * EH_ + en] = f2bf(leaky(acc[r] + be1));
            }
            __syncthreads();
            {   // L2: K = EH
                v8f acc = {};
                for (int ks = 0; ks < EH_ / 32; ++ks) {
                    v16bf a = load_frag_bf16(e1, EH_, 0, ks * 32, lane);
                    v16bf b = load_frag_bf16(A.Ehw2, EH_, wave * 16, ks * 32, lane);
                    acc = wmma_bf16(a, b, acc);
                }
                const int rb = 8 * (lane >> 4);
#pragma unroll
                for (int r = 0; r < 8; ++r) e2[(rb + r) * EH_ + en] = f2bf(leaky(acc[r] + be2));
            }
            __syncthreads();
            // L3: alpha = exp(.), write scaled h (bf16) for the hh GEMMs
            for (int j = 0; j < 4; ++j) {
                const int nt = wave + 8 * j;
                v8f acc = {};
                for (int ks = 0; ks < EH_ / 32; ++ks) {
                    v16bf a = load_frag_bf16(e2, EH_, 0, ks * 32, lane);
                    v16bf b = load_frag_bf16(A.Ehw3, EH_, nt * 16, ks * 32, lane);
                    acc = wmma_bf16(a, b, acc);
                }
                const int n = nt * 16 + (lane & 15);
                const int rb = 8 * (lane >> 4);
#pragma unroll
                for (int r = 0; r < 8; ++r) {
                    const int gr = row0 + rb + r;          // 0..127 = l*B + b
                    const float alpha = __expf(acc[r] + be3[j]);
                    A.hsc[(size_t)gr * H_ + n] = f2bf(bf2f(hT[(rb + r) * H_ + n]) * alpha);
                }
            }
        }
        gbar(A.bar, epoch);

        // ---- TDM #2/#3: DMA both h_scaled tiles (now globally visible).
        // They complete in issue order behind the x tile, so the waits below
        // can be incremental, overlapping each DMA with a live GEMM.
        if (wave == 0) {
            tdm_load_tile_bf16(sh0_lds, A.hsc + (size_t)mt * 16 * H_, H_, 16, H_);
            tdm_load_tile_bf16(sh1_lds, A.hsc + (size_t)(B_ + mt * 16) * H_, H_, 16, H_);
        }

        // ---- Phase B: layer 0
        __builtin_amdgcn_s_wait_tensorcnt(2);   // x tile landed
        __syncthreads();
        v8f accB[4] = {};
        gemm_quad(accB, xtile, 0, I_, A.Wih0, ht, lane);      // ih (LDS A)
        __builtin_amdgcn_s_wait_tensorcnt(1);   // h_scaled[0] tile landed
        __syncthreads();
        gemm_quad(accB, sh0, 0, H_, A.Whh0, ht, lane);        // hh (LDS A)
        cell_update(accB, bgB, t, mt, ht, lane,
                    A.h_state, A.c_state, A.h0n, nullptr);
        gbar(A.bar, epoch);

        // ---- Phase C: layer 1 (input = layer-0 new h, unscaled)
        __builtin_amdgcn_s_wait_tensorcnt(0);   // h_scaled[1] tile landed
        __syncthreads();
        v8f accC[4] = {};
        gemm_quad(accC, A.h0n, mt * 16, H_, A.Wih1, ht, lane); // ih (global A)
        gemm_quad(accC, sh1, 0, H_, A.Whh1, ht, lane);         // hh (LDS A)
        cell_update(accC, bgC, t, mt, ht, lane,
                    A.h_state + (size_t)B_ * H_, A.c_state + (size_t)B_ * H_,
                    nullptr, A.out);
        gbar(A.bar, epoch);
    }
}

// -------------------------------------------------------------- launcher ----
extern "C" void kernel_launch(void* const* d_in, const int* in_sizes, int n_in,
                              void* d_out, int out_size, void* d_ws, size_t ws_size,
                              hipStream_t stream) {
    (void)in_sizes; (void)n_in; (void)out_size; (void)ws_size;
    const float* input = (const float*)d_in[0];
    const float* h0    = (const float*)d_in[1];
    const float* c0    = (const float*)d_in[2];
    const float* Wih0f = (const float*)d_in[3];
    const float* Whh0f = (const float*)d_in[4];
    const float* bih0  = (const float*)d_in[5];
    const float* bhh0  = (const float*)d_in[6];
    const float* Wih1f = (const float*)d_in[7];
    const float* Whh1f = (const float*)d_in[8];
    const float* bih1  = (const float*)d_in[9];
    const float* bhh1  = (const float*)d_in[10];
    const float* eyw1f = (const float*)d_in[11];
    const float* eyb1  = (const float*)d_in[12];
    const float* eyw2f = (const float*)d_in[13];
    const float* eyb2  = (const float*)d_in[14];
    const float* eyw3f = (const float*)d_in[15];
    const float* eyb3  = (const float*)d_in[16];
    const float* ehw1f = (const float*)d_in[17];
    const float* ehb1  = (const float*)d_in[18];
    const float* ehw2f = (const float*)d_in[19];
    const float* ehb2  = (const float*)d_in[20];
    const float* ehw3f = (const float*)d_in[21];
    const float* ehb3  = (const float*)d_in[22];

    // workspace carve (≈75 MB total)
    uint8_t* base = (uint8_t*)d_ws;
    size_t off = 0;
    auto carve = [&](size_t bytes) -> void* {
        void* p = base + off;
        off += bytes;
        off = (off + 255) & ~(size_t)255;
        return p;
    };
    unsigned* bar = (unsigned*)carve(sizeof(unsigned));
    bf16_t* Wih0 = (bf16_t*)carve((size_t)4 * H_ * I_ * 2);
    bf16_t* Whh0 = (bf16_t*)carve((size_t)4 * H_ * H_ * 2);
    bf16_t* Wih1 = (bf16_t*)carve((size_t)4 * H_ * H_ * 2);
    bf16_t* Whh1 = (bf16_t*)carve((size_t)4 * H_ * H_ * 2);
    bf16_t* Eyw1 = (bf16_t*)carve((size_t)EH_ * I_ * 2);
    bf16_t* Eyw2 = (bf16_t*)carve((size_t)EH_ * EH_ * 2);
    bf16_t* Eyw3 = (bf16_t*)carve((size_t)I_ * EH_ * 2);
    bf16_t* Ehw1 = (bf16_t*)carve((size_t)EH_ * H_ * 2);
    bf16_t* Ehw2 = (bf16_t*)carve((size_t)EH_ * EH_ * 2);
    bf16_t* Ehw3 = (bf16_t*)carve((size_t)H_ * EH_ * 2);
    bf16_t* xs   = (bf16_t*)carve((size_t)T_ * B_ * I_ * 2);
    bf16_t* hsc  = (bf16_t*)carve((size_t)L_ * B_ * H_ * 2);
    bf16_t* h0n  = (bf16_t*)carve((size_t)B_ * H_ * 2);

    float* out     = (float*)d_out;
    float* h_state = out + (size_t)B_ * T_ * H_;       // h_n region doubles as state
    float* c_state = h_state + (size_t)L_ * B_ * H_;   // c_n region doubles as state

    k_init<<<64, 256, 0, stream>>>(h0, c0, h_state, c_state, bar);

    CvtArgs ca;
    ca.src[0] = Wih0f; ca.dst[0] = Wih0; ca.n[0] = 4 * H_ * I_;
    ca.src[1] = Whh0f; ca.dst[1] = Whh0; ca.n[1] = 4 * H_ * H_;
    ca.src[2] = Wih1f; ca.dst[2] = Wih1; ca.n[2] = 4 * H_ * H_;
    ca.src[3] = Whh1f; ca.dst[3] = Whh1; ca.n[3] = 4 * H_ * H_;
    ca.src[4] = eyw1f; ca.dst[4] = Eyw1; ca.n[4] = EH_ * I_;
    ca.src[5] = eyw2f; ca.dst[5] = Eyw2; ca.n[5] = EH_ * EH_;
    ca.src[6] = eyw3f; ca.dst[6] = Eyw3; ca.n[6] = I_ * EH_;
    ca.src[7] = ehw1f; ca.dst[7] = Ehw1; ca.n[7] = EH_ * H_;
    ca.src[8] = ehw2f; ca.dst[8] = Ehw2; ca.n[8] = EH_ * EH_;
    ca.src[9] = ehw3f; ca.dst[9] = Ehw3; ca.n[9] = H_ * EH_;
    k_cvt<<<512, 256, 0, stream>>>(ca);

    k_ency<<<(B_ * T_) / 16, 128, 0, stream>>>(input, eyb1, eyb2, eyb3,
                                               Eyw1, Eyw2, Eyw3, xs);

    RnnArgs ra;
    ra.bar = bar; ra.xs = xs; ra.hsc = hsc; ra.h0n = h0n;
    ra.Wih0 = Wih0; ra.Whh0 = Whh0; ra.Wih1 = Wih1; ra.Whh1 = Whh1;
    ra.Ehw1 = Ehw1; ra.Ehw2 = Ehw2; ra.Ehw3 = Ehw3;
    ra.bih0 = bih0; ra.bhh0 = bhh0; ra.bih1 = bih1; ra.bhh1 = bhh1;
    ra.ehb1 = ehb1; ra.ehb2 = ehb2; ra.ehb3 = ehb3;
    ra.out = out; ra.h_state = h_state; ra.c_state = c_state;
    k_rnn<<<NBLK, 256, 0, stream>>>(ra);
}
